// Net_51857435132346
// MI455X (gfx1250) — matrix operand order repaired and checked by reading, not compiled, not run on hardware
//
#include <hip/hip_runtime.h>

// ---------------------------------------------------------------------------
// MI455X (gfx1250) implementation.
// Dominant cost: fused 190-channel 7x7 conv (312 GFLOP) -> implicit GEMM on
// v_wmma_f32_16x16x32_bf16 (K=32 = input channels, one WMMA per tap).
// Round 2: register-block N (2 ntiles x 2 rows per wave) to cut L2 weight
// traffic 4x (19.7GB -> 4.9GB), and stage the LDS halo with
// global_load_async_to_lds_b128 (ASYNCcnt) instead of load+ds_store.
// ---------------------------------------------------------------------------

typedef __attribute__((ext_vector_type(4)))  unsigned int u32x4;
typedef __attribute__((ext_vector_type(16))) __bf16       bf16x16;
typedef __attribute__((ext_vector_type(8)))  float        f32x8;

#define BB   2
#define HH   512
#define WW   512
#define NPIX (BB * HH * WW)
#define EPSF 1e-5f
#define MCH  190   // 16+25+36+49+64 concatenated branch channels
#define MT   12    // M tiles of 16 (padded to 192)
#define TAPS 49

struct Params { const float* p[57]; };

union BF16V { u32x4 q[2]; bf16x16 v; };

static __device__ __forceinline__ unsigned short f2bf(float f) {
  union { float f; unsigned u; } v; v.f = f;
  unsigned u = v.u;
  unsigned r = u + 0x7FFFu + ((u >> 16) & 1u);   // round-to-nearest-even
  return (unsigned short)(r >> 16);
}

// ---------------------------------------------------------------------------
// Kernel 1: t = bn_init(x2 - x1)   (scalar-channel affine)
// ---------------------------------------------------------------------------
__global__ void k_diff(Params P, float* __restrict__ t) {
  int i = blockIdx.x * blockDim.x + threadIdx.x;
  if (i >= NPIX) return;
  float g = P.p[2][0], b = P.p[3][0], m = P.p[4][0], v = P.p[5][0];
  float a0 = g / sqrtf(v + EPSF);
  float d  = P.p[1][i] - P.p[0][i];
  t[i] = (d - m) * a0 + b;
}

// ---------------------------------------------------------------------------
// Kernel 2: X32 = relu(bn(conv7x7_1->32(t)))  stored bf16 channel-major [pix][32]
// ---------------------------------------------------------------------------
__global__ void k_conv1(Params P, const float* __restrict__ t,
                        u32x4* __restrict__ X32q) {
  int p = blockIdx.x * blockDim.x + threadIdx.x;
  if (p >= NPIX) return;
  int b   = p / (HH * WW);
  int rem = p % (HH * WW);
  int y = rem / WW, x = rem % WW;
  const float* w1 = P.p[6];
  float acc[32];
#pragma unroll
  for (int o = 0; o < 32; o++) acc[o] = 0.f;
  for (int ky = 0; ky < 7; ky++) {
    int yy = y + ky - 3;
    bool yok = (yy >= 0) && (yy < HH);
    for (int kx = 0; kx < 7; kx++) {
      int xx = x + kx - 3;
      float v = 0.f;
      if (yok && xx >= 0 && xx < WW) v = t[(b * HH + yy) * WW + xx];
      int tap = ky * 7 + kx;
#pragma unroll
      for (int o = 0; o < 32; o++) acc[o] += w1[o * 49 + tap] * v;
    }
  }
  unsigned d[16];
#pragma unroll
  for (int j = 0; j < 16; j++) {
    unsigned lo, hi;
#pragma unroll
    for (int h = 0; h < 2; h++) {
      int o = 2 * j + h;
      float g = P.p[7][o], bb2 = P.p[8][o], mm = P.p[9][o], vv = P.p[10][o];
      float A = g / sqrtf(vv + EPSF);
      float val = fmaxf(acc[o] * A + (bb2 - mm * A), 0.f);
      unsigned u = f2bf(val);
      if (h == 0) lo = u; else hi = u;
    }
    d[j] = lo | (hi << 16);
  }
  u32x4* dst = X32q + (size_t)p * 4;
#pragma unroll
  for (int i = 0; i < 4; i++) {
    u32x4 q;
    q[0] = d[4 * i]; q[1] = d[4 * i + 1]; q[2] = d[4 * i + 2]; q[3] = d[4 * i + 3];
    dst[i] = q;
  }
}

// ---------------------------------------------------------------------------
// Kernel 3: pack conv2 weights (BN scale folded) into WMMA A-matrix lane
// layout (ISA 7.12.2, 16-bit A 16x32) + per-channel bias/coef + constant C0.
// ---------------------------------------------------------------------------
__global__ void k_prep(Params P, unsigned* __restrict__ wpack,
                       float* __restrict__ bias, float* __restrict__ coef,
                       float* __restrict__ c0) {
  const int off[6] = {0, 16, 41, 77, 126, 190};
  const float* dw[5] = {P.p[11], P.p[16], P.p[21], P.p[26], P.p[31]};
  const float* dg[5] = {P.p[12], P.p[17], P.p[22], P.p[27], P.p[32]};
  const float* db[5] = {P.p[13], P.p[18], P.p[23], P.p[28], P.p[33]};
  const float* dm[5] = {P.p[14], P.p[19], P.p[24], P.p[29], P.p[34]};
  const float* dv[5] = {P.p[15], P.p[20], P.p[25], P.p[30], P.p[35]};
  const float* wsw = P.p[56];
  int tid = blockIdx.x * blockDim.x + threadIdx.x;

  int nslots = MT * TAPS * 32;
  if (tid < nslots) {
    int mtile = tid / (TAPS * 32);
    int rem   = tid % (TAPS * 32);
    int tap   = rem / 32;
    int lane  = rem % 32;
    int mloc  = lane & 15;
    int half8 = (lane >> 4) * 8;
    int mg = mtile * 16 + mloc;
    int br = -1, o = 0;
    if (mg < MCH) {
      for (int i = 0; i < 5; i++)
        if (mg >= off[i] && mg < off[i + 1]) { br = i; o = mg - off[i]; }
    }
    float A = 0.f;
    if (br >= 0) A = dg[br][o] / sqrtf(dv[br][o] + EPSF);
    unsigned short h[16];
#pragma unroll
    for (int i = 0; i < 16; i++) {
      int j = i >> 1, hb = i & 1;
      int k = (j < 4) ? (half8 + j * 2 + hb) : (16 + half8 + (j - 4) * 2 + hb);
      float w = 0.f;
      if (br >= 0) w = dw[br][(o * 32 + k) * 49 + tap] * A;
      h[i] = f2bf(w);
    }
    unsigned base = (unsigned)tid * 8u;
#pragma unroll
    for (int j = 0; j < 8; j++)
      wpack[base + j] = (unsigned)h[2 * j] | ((unsigned)h[2 * j + 1] << 16);
  }

  if (tid < 192) {
    int m = tid;
    float bi = 0.f, cf = 0.f;
    if (m < MCH) {
      int br = 0, o = 0;
      for (int i = 0; i < 5; i++)
        if (m >= off[i] && m < off[i + 1]) { br = i; o = m - off[i]; }
      float A  = dg[br][o] / sqrtf(dv[br][o] + EPSF);
      bi = db[br][o] - dm[br][o] * A;
      float ug = P.p[36 + 4 * br][0], ub = P.p[37 + 4 * br][0];
      float um = P.p[38 + 4 * br][0], uv = P.p[39 + 4 * br][0];
      float uA = ug / sqrtf(uv + EPSF);
      float we = (br == 0) ? wsw[0] : fabsf(wsw[br]);
      cf = we * uA / (5.f * (float)(off[br + 1] - off[br]));
    }
    bias[m] = bi;
    coef[m] = cf;
  }
  if (tid == 0) {
    float s = 0.f;
    for (int br = 0; br < 5; br++) {
      float ug = P.p[36 + 4 * br][0], ub = P.p[37 + 4 * br][0];
      float um = P.p[38 + 4 * br][0], uv = P.p[39 + 4 * br][0];
      float uA = ug / sqrtf(uv + EPSF);
      float uB = ub - um * uA;
      float we = (br == 0) ? wsw[0] : fabsf(wsw[br]);
      s += we * uB;
    }
    c0[0] = s / 5.f;
  }
}

// ---------------------------------------------------------------------------
// Kernel 4: fused 190-ch 7x7 conv via WMMA bf16. 128 threads = 4 waves, one
// M-group (3 M-tiles) per wave; each wave register-blocks 4 B-tiles
// (2 ntiles x 2 rows) per A load -> 12 WMMAs per tap, 4x A-traffic reuse.
// Halo staged to LDS with async-to-LDS loads (ASYNCcnt).
// ---------------------------------------------------------------------------
__global__ __launch_bounds__(128) void k_main(
    Params P, const u32x4* __restrict__ X32q, const u32x4* __restrict__ wpackq,
    const float* __restrict__ bias, const float* __restrict__ coef,
    const float* __restrict__ c0, float* __restrict__ out) {
  __shared__ u32x4 sX[8 * 38 * 4];      // 38 cols x 8 rows x 32ch bf16 halo
  __shared__ float psumArr[2 * 4 * 32]; // [row][mgroup][px]

  int tid  = threadIdx.x;
  int wave = tid >> 5, lane = tid & 31;
  int mgroup = wave;                    // 4 waves = 4 M-groups
  int bx = blockIdx.x;                  // 0..15 (32-px column blocks)
  int r0 = blockIdx.y * 2;              // 0..1022, two rows per WG
  int b = r0 >> 9, y0 = r0 & 511;

  // ---- Stage halo tile of X32 into LDS via async-to-LDS (zero-pad borders).
  const unsigned long long gbase = (unsigned long long)X32q;
  for (int s = tid; s < 8 * 38; s += 128) {
    int r = s / 38, c = s % 38;
    int yy = y0 + r - 3;
    int xx = bx * 32 + c - 3;
    unsigned lds = (unsigned)(unsigned long long)(&sX[s * 4]);
    if (yy >= 0 && yy < HH && xx >= 0 && xx < WW) {
      unsigned voff = (unsigned)(((b * HH + yy) * WW + xx) * 64);
      asm volatile("global_load_async_to_lds_b128 %0, %1, %2 offset:0"
                   :: "v"(lds), "v"(voff), "s"(gbase) : "memory");
      asm volatile("global_load_async_to_lds_b128 %0, %1, %2 offset:16"
                   :: "v"(lds), "v"(voff), "s"(gbase) : "memory");
      asm volatile("global_load_async_to_lds_b128 %0, %1, %2 offset:32"
                   :: "v"(lds), "v"(voff), "s"(gbase) : "memory");
      asm volatile("global_load_async_to_lds_b128 %0, %1, %2 offset:48"
                   :: "v"(lds), "v"(voff), "s"(gbase) : "memory");
    } else {
      u32x4 z = {0u, 0u, 0u, 0u};
      sX[s * 4] = z; sX[s * 4 + 1] = z; sX[s * 4 + 2] = z; sX[s * 4 + 3] = z;
    }
  }
  asm volatile("s_wait_asynccnt 0" ::: "memory");
  __syncthreads();

  int n   = lane & 15;                  // pixel within N-tile
  int kh2 = (lane >> 4);                // K-quad select (u32x4 units): h, h+2
  int mt0 = mgroup * 3;
  const size_t astride = (size_t)TAPS * 32 * 2;   // one M-tile, u32x4 units

  f32x8 acc[4][3];                      // [ntile+row block][mtile]
#pragma unroll
  for (int nb = 0; nb < 4; nb++)
#pragma unroll
    for (int mt = 0; mt < 3; mt++)
#pragma unroll
      for (int r = 0; r < 8; r++) acc[nb][mt][r] = 0.f;

  for (int dy = 0; dy < 7; dy++) {
    for (int dx = 0; dx < 7; dx++) {
      int tap = dy * 7 + dx;
      size_t abase = ((size_t)(mt0 * TAPS + tap) * 32 + lane) * 2;
      BF16V a[3];
#pragma unroll
      for (int mt = 0; mt < 3; mt++) {
        a[mt].q[0] = wpackq[abase + (size_t)mt * astride];
        a[mt].q[1] = wpackq[abase + (size_t)mt * astride + 1];
      }
#pragma unroll
      for (int nb = 0; nb < 4; nb++) {
        int rowS = nb >> 1, nt = nb & 1;
        int slot = ((dy + rowS) * 38 + nt * 16 + n + dx) * 4;
        BF16V bv;
        bv.q[0] = sX[slot + kh2];
        bv.q[1] = sX[slot + kh2 + 2];
#pragma unroll
        for (int mt = 0; mt < 3; mt++)
          acc[nb][mt] = __builtin_amdgcn_wmma_f32_16x16x32_bf16(
              false, a[mt].v, false, bv.v, (short)0, acc[nb][mt], false, false);
      }
    }
  }

  // ---- Epilogue: relu(acc+bias)*coef over this wave's 48 channels.
  int half8 = (lane >> 4) * 8;
#pragma unroll
  for (int nb = 0; nb < 4; nb++) {
    float ps = 0.f;
#pragma unroll
    for (int mt = 0; mt < 3; mt++) {
#pragma unroll
      for (int r = 0; r < 8; r++) {
        int m = (mt0 + mt) * 16 + half8 + r;
        ps += fmaxf(acc[nb][mt][r] + bias[m], 0.f) * coef[m];
      }
    }
    ps += __shfl_xor(ps, 16, 32);       // fold the two lane halves
    int rowS = nb >> 1, nt = nb & 1;
    if (lane < 16) psumArr[rowS * 128 + mgroup * 32 + nt * 16 + n] = ps;
  }
  __syncthreads();

  if (tid < 64) {
    int rowS = tid >> 5, px = tid & 31;
    int base = rowS * 128 + px;
    float tot = psumArr[base] + psumArr[base + 32] + psumArr[base + 64] +
                psumArr[base + 96] + c0[0];
    int y = y0 + rowS;
    int x = bx * 32 + px;
    float di = (float)y - 256.0f, dj = (float)x - 256.0f;
    bool mask = (di * di + dj * dj) > 65537.0f;   // 256^2 + 1
    float o = mask ? 10.0f : tot;
    size_t idx = (size_t)(b * HH + y) * WW + x;
    out[idx] = P.p[1][idx] - o;
  }
}

// ---------------------------------------------------------------------------
extern "C" void kernel_launch(void* const* d_in, const int* in_sizes, int n_in,
                              void* d_out, int out_size, void* d_ws, size_t ws_size,
                              hipStream_t stream) {
  (void)in_sizes; (void)out_size; (void)ws_size;
  Params P;
  for (int i = 0; i < 57 && i < n_in; i++) P.p[i] = (const float*)d_in[i];

  char* ws = (char*)d_ws;
  // workspace layout (needs ~36.3 MB)
  size_t offX32  = 0;                                  // 16,777,216 bf16 = 33.5MB
  size_t offT    = offX32 + (size_t)NPIX * 32 * 2;
  size_t offW    = offT + (size_t)NPIX * 4;            // t f32 = 2MB
  size_t offBias = offW + (size_t)MT * TAPS * 32 * 8 * 4;  // wpack 588KB
  size_t offCoef = offBias + 192 * 4;
  size_t offC0   = offCoef + 192 * 4;

  u32x4*    X32q  = (u32x4*)(ws + offX32);
  float*    t     = (float*)(ws + offT);
  unsigned* wpack = (unsigned*)(ws + offW);
  float*    bias  = (float*)(ws + offBias);
  float*    coef  = (float*)(ws + offCoef);
  float*    c0    = (float*)(ws + offC0);

  k_diff <<<dim3((NPIX + 255) / 256), dim3(256), 0, stream>>>(P, t);
  k_conv1<<<dim3((NPIX + 255) / 256), dim3(256), 0, stream>>>(P, t, X32q);
  k_prep <<<dim3((MT * TAPS * 32 + 255) / 256), dim3(256), 0, stream>>>(
      P, wpack, bias, coef, c0);
  k_main <<<dim3(16, (BB * HH) / 2), dim3(128), 0, stream>>>(
      P, X32q, (const u32x4*)wpack, bias, coef, c0, (float*)d_out);
}
